// WaveletMoeTimeAttention_68135361184392
// MI455X (gfx1250) — matrix operand order repaired and checked by reading, not compiled, not run on hardware
//
#include <hip/hip_runtime.h>
#include <hip/hip_bf16.h>
#include <cstdint>
#include <cstddef>

// Model dims (fixed by the reference)
#define BB   2
#define SS   2048
#define HH   1024
#define NH   16
#define NKV  4
#define HD   64

typedef __attribute__((ext_vector_type(16))) __bf16 v16bf;
typedef __attribute__((ext_vector_type(8)))  float  v8f;
typedef __attribute__((ext_vector_type(4)))  int    v4i;

union Frag16 { v16bf v; unsigned short u[16]; };
struct __align__(16) U128 { unsigned int w[4]; };
union V128 { U128 q; unsigned short s[8]; };

#define AS1 __attribute__((address_space(1)))
#define AS3 __attribute__((address_space(3)))

#if __has_builtin(__builtin_amdgcn_global_load_async_to_lds_b128) && \
    __has_builtin(__builtin_amdgcn_s_wait_asynccnt)
#define HAVE_ASYNC_LDS 1
#else
#define HAVE_ASYNC_LDS 0
#endif

// 16-byte global -> LDS copy: async (ASYNCcnt) on CDNA5 when available.
__device__ __forceinline__ void lds_copy16(void* lds_dst, const void* gsrc) {
#if HAVE_ASYNC_LDS
  __builtin_amdgcn_global_load_async_to_lds_b128(
      (AS1 v4i*)gsrc, (AS3 v4i*)lds_dst, 0, 0);
#else
  *(U128*)lds_dst = *(const U128*)gsrc;
#endif
}
__device__ __forceinline__ void lds_copy_fence() {
#if HAVE_ASYNC_LDS
  __builtin_amdgcn_s_wait_asynccnt(0);
#endif
}

__device__ __forceinline__ unsigned short f2bf(float f) {
  unsigned int x = __float_as_uint(f);
  unsigned int r = ((x >> 16) & 1u) + 0x7FFFu;   // round-to-nearest-even
  return (unsigned short)((x + r) >> 16);
}

// ---------------------------------------------------------------------------
// Elementwise f32 -> bf16 cast (grid-stride)
// ---------------------------------------------------------------------------
__global__ void cast_bf16_kernel(const float* __restrict__ src,
                                 unsigned short* __restrict__ dst, int n) {
  for (int i = blockIdx.x * blockDim.x + threadIdx.x; i < n;
       i += gridDim.x * blockDim.x)
    dst[i] = f2bf(src[i]);
}

// ---------------------------------------------------------------------------
// RoPE + reshape [row, h*64+d] (f32) -> [b][h][s][d] (bf16).  rope==0: cast only.
// ---------------------------------------------------------------------------
__global__ void rope_cast_kernel(const float* __restrict__ src,
                                 unsigned short* __restrict__ dst,
                                 int nheads, int rope) {
  const int stride = nheads * HD;
  const int total  = BB * SS * nheads * HD;
  for (int idx = blockIdx.x * blockDim.x + threadIdx.x; idx < total;
       idx += gridDim.x * blockDim.x) {
    int d   = idx & (HD - 1);
    int t   = idx >> 6;
    int h   = t % nheads;
    int row = t / nheads;          // b*S + s
    int b   = row >> 11;
    int s   = row & (SS - 1);
    float val = src[row * stride + h * HD + d];
    if (rope) {
      int   j   = d & 31;
      float inv = __powf(10000.0f, -(float)j * (1.0f / 32.0f));
      float ang = (float)s * inv;
      float other = src[row * stride + h * HD + (d ^ 32)];
      float rot   = (d < 32) ? -other : other;
      val = val * __cosf(ang) + rot * __sinf(ang);
    }
    dst[(((size_t)b * nheads + h) * SS + s) * HD + d] = f2bf(val);
  }
}

// ---------------------------------------------------------------------------
// C[M,N] = A[M,K](bf16) * B[N,K]^T(bf16) + bias, fp32 out.
// Grid: (N/64, M/64), block 128 threads (4 waves).  K multiple of 64.
// K-stage depth 64 (two 16x16x32 WMMA steps per barrier), async LDS staging.
// ---------------------------------------------------------------------------
__global__ void gemm_bf16_nt(const unsigned short* __restrict__ A, int lda,
                             const unsigned short* __restrict__ Bw, int ldb,
                             const float* __restrict__ bias,
                             float* __restrict__ C, int ldc, int K) {
  __shared__ __align__(16) unsigned short As[64 * 64];
  __shared__ __align__(16) unsigned short Bs[64 * 64];

  const int tid  = threadIdx.x;
  const int lane = tid & 31;
  const int w    = tid >> 5;
  const int r    = lane & 15;
  const int hi   = lane >> 4;
  const int n0   = blockIdx.x * 64;
  const int m0   = blockIdx.y * 64;

  v8f acc[4] = {};

  for (int k0 = 0; k0 < K; k0 += 64) {
    // Stage 64x64 tiles of A and B (8-element 16B chunks, async to LDS).
    for (int c = tid; c < 512; c += 128) {
      int i = c >> 3, ko = (c & 7) * 8;
      lds_copy16(&As[i * 64 + ko], &A[(size_t)(m0 + i) * lda + k0 + ko]);
      lds_copy16(&Bs[i * 64 + ko], &Bw[(size_t)(n0 + i) * ldb + k0 + ko]);
    }
    lds_copy_fence();
    __syncthreads();

#pragma unroll
    for (int kk = 0; kk < 64; kk += 32) {
      Frag16 a;
      const unsigned short* ar = &As[(16 * w + r) * 64 + kk];
#pragma unroll
      for (int e = 0; e < 8; ++e) {
        a.u[e]     = ar[8 * hi + e];
        a.u[8 + e] = ar[16 + 8 * hi + e];
      }
      Frag16 bf[4];
#pragma unroll
      for (int j = 0; j < 4; ++j) {
        const unsigned short* br = &Bs[(16 * j + r) * 64 + kk];
#pragma unroll
        for (int e = 0; e < 8; ++e) { bf[j].u[e] = br[8 * hi + e]; bf[j].u[8 + e] = br[16 + 8 * hi + e]; }
      }
#pragma unroll
      for (int j = 0; j < 4; ++j)
        acc[j] = __builtin_amdgcn_wmma_f32_16x16x32_bf16(
            false, a.v, false, bf[j].v, (short)0, acc[j], false, false);
    }
    __syncthreads();
  }

#pragma unroll
  for (int j = 0; j < 4; ++j) {
    int col = n0 + 16 * j + r;
    float bvv = bias ? bias[col] : 0.0f;
#pragma unroll
    for (int rr = 0; rr < 8; ++rr) {
      int row = m0 + 16 * w + rr + 8 * hi;
      C[(size_t)row * ldc + col] = acc[j][rr] + bvv;
    }
  }
}

// ---------------------------------------------------------------------------
// Flash attention.  Block = (b, h, 64 query rows), 128 threads (4 waves,
// 16 query rows per wave).  32-key steps; bf16 WMMA, fp32 online softmax.
// Writes attn output in [b*S, NH*HD] bf16 layout for the Wo GEMM.
// ---------------------------------------------------------------------------
__global__ void attn_flash_kernel(const unsigned short* __restrict__ Qbf,
                                  const unsigned short* __restrict__ Kbf,
                                  const unsigned short* __restrict__ Vbf,
                                  unsigned short* __restrict__ Obf) {
  __shared__ __align__(16) unsigned short kt[32 * 64];      // [key][d]
  __shared__ __align__(16) unsigned short vt[64 * 32];      // [d][key] (V^T)
  __shared__ __align__(16) unsigned short pbuf[4][16 * 32]; // per-wave P tile

  const int tid  = threadIdx.x;
  const int lane = tid & 31;
  const int w    = tid >> 5;
  const int r    = lane & 15;
  const int hi   = lane >> 4;

  const int bid = blockIdx.x;
  const int qt  = bid & 31;          // S/64 = 32 query tiles
  const int h   = (bid >> 5) & (NH - 1);
  const int b   = bid >> 9;
  const int kvh = h >> 2;            // NH/NKV = 4
  const int qb0 = qt * 64;

  // Load this wave's Q fragments (16 rows x 64 dims -> two A fragments).
  const unsigned short* qrow =
      Qbf + ((size_t)(b * NH + h) * SS + qb0 + 16 * w + r) * HD;
  Frag16 qa[2];
#pragma unroll
  for (int p = 0; p < 2; ++p)
#pragma unroll
    for (int e = 0; e < 8; ++e) {
      qa[p].u[e]     = qrow[32 * p + 8 * hi + e];
      qa[p].u[8 + e] = qrow[32 * p + 16 + 8 * hi + e];
    }

  v8f   acc[4] = {};
  float Mrow[8], Lrow[8];
#pragma unroll
  for (int rr = 0; rr < 8; ++rr) { Mrow[rr] = -1e30f; Lrow[rr] = 0.0f; }

  const unsigned short* kg = Kbf + (size_t)(b * NKV + kvh) * SS * HD;
  const unsigned short* vg = Vbf + (size_t)(b * NKV + kvh) * SS * HD;
  const int kend = qb0 + 64;         // causal upper bound for this block

  for (int k0 = 0; k0 < kend; k0 += 32) {
    // Stage K tile [32][64] (async) and transposed V tile [64][32] (manual).
    for (int c = tid; c < 256; c += 128) {
      int i = c >> 3, doff = (c & 7) * 8;
      lds_copy16(&kt[i * 64 + doff], &kg[(size_t)(k0 + i) * HD + doff]);
      V128 t;
      t.q = *(const U128*)&vg[(size_t)(k0 + i) * HD + doff];
#pragma unroll
      for (int e = 0; e < 8; ++e) vt[(doff + e) * 32 + i] = t.s[e];
    }
    lds_copy_fence();
    __syncthreads();

    // S(16x32) = Q(16x64) * K_tile(64x32)^T : two C frags, two K-depth steps.
    v8f sc[2] = {};
#pragma unroll
    for (int j = 0; j < 2; ++j)
#pragma unroll
      for (int p = 0; p < 2; ++p) {
        Frag16 kb;
        const unsigned short* kr = &kt[(16 * j + r) * 64 + 32 * p];
#pragma unroll
        for (int e = 0; e < 8; ++e) { kb.u[e] = kr[8 * hi + e]; kb.u[8 + e] = kr[16 + 8 * hi + e]; }
        sc[j] = __builtin_amdgcn_wmma_f32_16x16x32_bf16(
            false, qa[p].v, false, kb.v, (short)0, sc[j], false, false);
      }

    // Causal mask + online softmax (rows live across 16-lane N groups).
    const int key0 = k0 + r, key1 = k0 + 16 + r;
#pragma unroll
    for (int rr = 0; rr < 8; ++rr) {
      int qr = qb0 + 16 * w + rr + 8 * hi;
      float s0 = sc[0][rr] * 0.125f + ((key0 > qr) ? -1e9f : 0.0f);
      float s1 = sc[1][rr] * 0.125f + ((key1 > qr) ? -1e9f : 0.0f);
      float mx = fmaxf(s0, s1);
      mx = fmaxf(mx, __shfl_xor(mx, 1, 32));
      mx = fmaxf(mx, __shfl_xor(mx, 2, 32));
      mx = fmaxf(mx, __shfl_xor(mx, 4, 32));
      mx = fmaxf(mx, __shfl_xor(mx, 8, 32));
      float mnew = fmaxf(Mrow[rr], mx);
      float sc_  = __expf(Mrow[rr] - mnew);
      Mrow[rr] = mnew;
      float p0 = __expf(s0 - mnew);
      float p1 = __expf(s1 - mnew);
      float ps = p0 + p1;
      ps += __shfl_xor(ps, 1, 32);
      ps += __shfl_xor(ps, 2, 32);
      ps += __shfl_xor(ps, 4, 32);
      ps += __shfl_xor(ps, 8, 32);
      Lrow[rr] = Lrow[rr] * sc_ + ps;
      int m = rr + 8 * hi;
      pbuf[w][m * 32 + r]      = f2bf(p0);
      pbuf[w][m * 32 + 16 + r] = f2bf(p1);
#pragma unroll
      for (int j = 0; j < 4; ++j) acc[j][rr] = acc[j][rr] * sc_;
    }

    // Re-layout P (C-frag -> A-frag) via per-wave LDS, then O += P * V.
    Frag16 pa;
    const unsigned short* pr = &pbuf[w][r * 32];
#pragma unroll
    for (int e = 0; e < 8; ++e) {
      pa.u[e]     = pr[8 * hi + e];
      pa.u[8 + e] = pr[16 + 8 * hi + e];
    }
#pragma unroll
    for (int j = 0; j < 4; ++j) {
      Frag16 vb;
      const unsigned short* vr = &vt[(16 * j + r) * 32];
#pragma unroll
      for (int e = 0; e < 8; ++e) { vb.u[e] = vr[8 * hi + e]; vb.u[8 + e] = vr[16 + 8 * hi + e]; }
      acc[j] = __builtin_amdgcn_wmma_f32_16x16x32_bf16(
          false, pa.v, false, vb.v, (short)0, acc[j], false, false);
    }
    __syncthreads();
  }

  // Normalize and emit bf16 attn output at [b*S + s][h*64 + d].
#pragma unroll
  for (int j = 0; j < 4; ++j)
#pragma unroll
    for (int rr = 0; rr < 8; ++rr) {
      int row = qb0 + 16 * w + rr + 8 * hi;
      int col = h * HD + 16 * j + r;
      float v = acc[j][rr] / Lrow[rr];
      Obf[((size_t)b * SS + row) * HH + col] = f2bf(v);
    }
}

// ---------------------------------------------------------------------------
// Host orchestration
// ---------------------------------------------------------------------------
extern "C" void kernel_launch(void* const* d_in, const int* in_sizes, int n_in,
                              void* d_out, int out_size, void* d_ws,
                              size_t ws_size, hipStream_t stream) {
  (void)in_sizes; (void)n_in; (void)out_size; (void)ws_size;
  const float* hs = (const float*)d_in[0];
  // d_in[1] = attention_mask (causal; recomputed), d_in[2] = position_ids (= arange)
  const float* Wq = (const float*)d_in[3];
  const float* bq = (const float*)d_in[4];
  const float* Wk = (const float*)d_in[5];
  const float* bk = (const float*)d_in[6];
  const float* Wv = (const float*)d_in[7];
  const float* bv = (const float*)d_in[8];
  const float* Wo = (const float*)d_in[9];
  float* out = (float*)d_out;

  char* ws = (char*)d_ws;
  size_t off = 0;
  auto alloc = [&](size_t nbytes) -> void* {
    void* p = ws + off;
    off = (off + nbytes + 255) & ~(size_t)255;
    return p;
  };

  const size_t M = (size_t)BB * SS;                        // 4096 rows
  unsigned short* Xbf = (unsigned short*)alloc(M * HH * 2);
  unsigned short* Wqb = (unsigned short*)alloc((size_t)HH * HH * 2);
  unsigned short* Wkb = (unsigned short*)alloc((size_t)NKV * HD * HH * 2);
  unsigned short* Wvb = (unsigned short*)alloc((size_t)NKV * HD * HH * 2);
  unsigned short* Wob = (unsigned short*)alloc((size_t)HH * HH * 2);
  float* Qf = (float*)alloc(M * HH * 4);
  float* Kf = (float*)alloc(M * NKV * HD * 4);
  float* Vf = (float*)alloc(M * NKV * HD * 4);
  unsigned short* Qbf = (unsigned short*)alloc((size_t)BB * NH  * SS * HD * 2);
  unsigned short* Kbf = (unsigned short*)alloc((size_t)BB * NKV * SS * HD * 2);
  unsigned short* Vbf = (unsigned short*)alloc((size_t)BB * NKV * SS * HD * 2);
  unsigned short* Abf = (unsigned short*)alloc(M * HH * 2);

  // 1) bf16 casts of activations and weights.
  cast_bf16_kernel<<<dim3(2048), dim3(256), 0, stream>>>(hs, Xbf, (int)(M * HH));
  cast_bf16_kernel<<<dim3(1024), dim3(256), 0, stream>>>(Wq, Wqb, HH * HH);
  cast_bf16_kernel<<<dim3(256),  dim3(256), 0, stream>>>(Wk, Wkb, NKV * HD * HH);
  cast_bf16_kernel<<<dim3(256),  dim3(256), 0, stream>>>(Wv, Wvb, NKV * HD * HH);
  cast_bf16_kernel<<<dim3(1024), dim3(256), 0, stream>>>(Wo, Wob, HH * HH);

  // 2) Q/K/V projections (WMMA GEMMs, bias fused).
  gemm_bf16_nt<<<dim3(HH / 64, M / 64), dim3(128), 0, stream>>>(
      Xbf, HH, Wqb, HH, bq, Qf, HH, HH);
  gemm_bf16_nt<<<dim3((NKV * HD) / 64, M / 64), dim3(128), 0, stream>>>(
      Xbf, HH, Wkb, HH, bk, Kf, NKV * HD, HH);
  gemm_bf16_nt<<<dim3((NKV * HD) / 64, M / 64), dim3(128), 0, stream>>>(
      Xbf, HH, Wvb, HH, bv, Vf, NKV * HD, HH);

  // 3) RoPE (Q, K) and reshape/cast to [b][h][s][d] bf16; V cast only.
  rope_cast_kernel<<<dim3(4096), dim3(256), 0, stream>>>(Qf, Qbf, NH, 1);
  rope_cast_kernel<<<dim3(1024), dim3(256), 0, stream>>>(Kf, Kbf, NKV, 1);
  rope_cast_kernel<<<dim3(1024), dim3(256), 0, stream>>>(Vf, Vbf, NKV, 0);

  // 4) Flash attention: B*NH*(S/64) blocks.
  attn_flash_kernel<<<dim3(BB * NH * (SS / 64)), dim3(128), 0, stream>>>(
      Qbf, Kbf, Vbf, Abf);

  // 5) Output projection into d_out (fp32).
  gemm_bf16_nt<<<dim3(HH / 64, M / 64), dim3(128), 0, stream>>>(
      Abf, HH, Wob, HH, nullptr, out, HH, HH);
}
